// TransImg2_90658169684631
// MI455X (gfx1250) — matrix operand order repaired and checked by reading, not compiled, not run on hardware
//
#include <hip/hip_runtime.h>
#include <hip/hip_bf16.h>
#include <cstdint>
#include <math.h>

// ---------------------------------------------------------------------------
// TransformerConv pipeline for gfx1250 (MI455X), round 3.
// - fused q/k/v/s projection as one bf16 WMMA GEMM (v_wmma_f32_16x16x32_bf16)
// - A tiles: GLOBAL_LOAD_ASYNC_TO_LDS_B128 + double-buffered LDS (ASYNCcnt)
// - B fragments loaded straight from global (contiguous in [dout][din] layout)
// - all GEMM operands zero-padded in workspace -> unguarded 128-bit loads
// ---------------------------------------------------------------------------

#define NN 30000
#define EE 180000
#define MPAD 30080  // ceil(NN/128)*128 : padded row count for GEMM A operands

typedef __attribute__((ext_vector_type(16))) __bf16 v16bf;
typedef __attribute__((ext_vector_type(8)))  float  v8f;

// ------------------------------ helpers ------------------------------------

__device__ __forceinline__ void atomic_add_f32(float* p, float v) {
  __hip_atomic_fetch_add(p, v, __ATOMIC_RELAXED, __HIP_MEMORY_SCOPE_AGENT);
}

// order-preserving float -> u32 key (for atomic segment-max)
__device__ __forceinline__ unsigned fkey(float f) {
  unsigned u = __float_as_uint(f);
  return (u & 0x80000000u) ? ~u : (u | 0x80000000u);
}
__device__ __forceinline__ float funkey(unsigned k) {
  unsigned u = (k & 0x80000000u) ? (k & 0x7FFFFFFFu) : ~k;
  return __uint_as_float(u);
}

// async copy of 32 bytes/lane (global -> LDS), two B128 ops, ASYNCcnt +2.
// NOTE: INST_OFFSET is added to BOTH the LDS and global address (ISA 10.7),
// which matches a contiguous 32B/lane transfer.
__device__ __forceinline__ void async_copy32(unsigned lds_addr,
                                             const __bf16* gptr) {
  unsigned long long ga = (unsigned long long)(size_t)gptr;
  asm volatile(
      "global_load_async_to_lds_b128 %0, %1, off\n\t"
      "global_load_async_to_lds_b128 %0, %1, off offset:16"
      :: "v"(lds_addr), "v"(ga) : "memory");
}

// ------------------------------ packing casts ------------------------------

// f32 [Nrows, K] -> bf16 [MPAD, ldp], zero-filling column pad and row pad
__global__ __launch_bounds__(256) void cast_pad_bf16(
    const float* __restrict__ in, __bf16* __restrict__ out,
    int Nrows, int K, int ldp, long total) {
  long i = (long)blockIdx.x * blockDim.x + threadIdx.x;
  if (i >= total) return;
  int r = (int)(i / ldp);
  int c = (int)(i % ldp);
  __bf16 v = (__bf16)0.0f;
  if (r < Nrows && c < K) v = (__bf16)in[(long)r * K + c];
  out[i] = v;
}

// pack {Wq,Wk,Wv,Ws} ([dout,din] f32 each) into bf16 [Jpad, ldw], zero pad.
// row jj = p*dout + j keeps each weight's native row layout (K contiguous).
__global__ __launch_bounds__(256) void pack_weights(
    const float* __restrict__ W0, const float* __restrict__ W1,
    const float* __restrict__ W2, const float* __restrict__ W3,
    __bf16* __restrict__ Wp, int dout, int din, int ldw, long total) {
  long i = (long)blockIdx.x * blockDim.x + threadIdx.x;
  if (i >= total) return;
  int jj = (int)(i / ldw);
  int k  = (int)(i % ldw);
  __bf16 v = (__bf16)0.0f;
  if (k < din && jj < 4 * dout) {
    int p = jj / dout, j = jj % dout;
    const float* W = (p == 0) ? W0 : (p == 1) ? W1 : (p == 2) ? W2 : W3;
    v = (__bf16)W[(long)j * din + k];
  }
  Wp[i] = v;
}

__global__ __launch_bounds__(256) void pack_bias(
    const float* __restrict__ b0, const float* __restrict__ b1,
    const float* __restrict__ b2, const float* __restrict__ b3,
    float* __restrict__ bp, int dout, int Jpad) {
  int jj = blockIdx.x * blockDim.x + threadIdx.x;
  if (jj >= Jpad) return;
  float v = 0.f;
  if (jj < 4 * dout) {
    int p = jj / dout, j = jj % dout;
    const float* b = (p == 0) ? b0 : (p == 1) ? b1 : (p == 2) ? b2 : b3;
    v = b[j];
  }
  bp[jj] = v;
}

// ------------------------------ WMMA GEMM ----------------------------------
// C[M, Nout] = X[., ldx] @ Wp[Jpad, ldw]^T + bias.   X padded (rows>=grid*128,
// cols K..Kpad-1 zero), Wp padded. Kpad%32==0, ldx%8==0, ldw%32==0.
// Block 256 thr = 8 waves; macro tile 128x64; wave owns 2x2 of 16x16 tiles.

__global__ __launch_bounds__(256) void gemm_bf16_wmma(
    const __bf16* __restrict__ X, int ldx,
    const __bf16* __restrict__ Wp, int ldw,
    const float*  __restrict__ bias,
    float* __restrict__ C,
    int M, int Nout, int Kpad) {
  __shared__ __bf16 lA[2][128][40];  // double-buffered 128x32, 80B row pitch

  const int tid  = threadIdx.x;
  const int lane = tid & 31;
  const int wave = tid >> 5;
  const int wm   = wave >> 1;    // 0..3
  const int wn   = wave & 1;     // 0..1
  const long blockRow = (long)blockIdx.y * 128;
  const int  blockCol = blockIdx.x * 64;

  const int mL    = lane & 15;
  const int khalf = (lane >> 4) * 16;  // A: lanes 16-31 take K+16
  const int kb0   = (lane >> 4) * 16;  // B: lanes 16-31 take K+16

  // cooperative A-copy coords: thread -> (row, 16-elem half)
  const int arow = tid >> 1;
  const int acol = (tid & 1) * 16;
  const __bf16* aptr = X + (blockRow + arow) * ldx + acol;
  const unsigned ldsA[2] = {
      (unsigned)(size_t)&lA[0][arow][acol],
      (unsigned)(size_t)&lA[1][arow][acol]};

  // B fragment row pointers (column j of the packed weight = row of Wp)
  const __bf16* wrow[2];
#pragma unroll
  for (int sn = 0; sn < 2; ++sn)
    wrow[sn] = Wp + (long)(blockCol + wn * 32 + sn * 16 + mL) * ldw + kb0;

  v8f acc[2][2];
#pragma unroll
  for (int a = 0; a < 2; ++a)
#pragma unroll
    for (int b = 0; b < 2; ++b)
      acc[a][b] = (v8f){0.f, 0.f, 0.f, 0.f, 0.f, 0.f, 0.f, 0.f};

  const int nIter = Kpad >> 5;
  // prologue: tile 0 -> buffer 0
  async_copy32(ldsA[0], aptr);

  for (int it = 0; it < nIter; ++it) {
    const int cur = it & 1;
    const int k0  = it << 5;
    if (it + 1 < nIter) {
      // buffer (1-cur)'s readers finished at the end of iteration it-1
      async_copy32(ldsA[1 - cur], aptr + ((it + 1) << 5));
      asm volatile("s_wait_asynccnt 0x2" ::: "memory");  // tile 'cur' done
      __builtin_prefetch(wrow[0] + k0 + 32, 0, 3);       // next B (near)
    } else {
      asm volatile("s_wait_asynccnt 0x0" ::: "memory");
    }
    __syncthreads();  // everyone's slice of tile 'cur' visible

    // A fragments: two contiguous 16B runs per lane (ds_load_b128 x2)
    union { v16bf v; uint4 u[2]; } af[2];
#pragma unroll
    for (int sm = 0; sm < 2; ++sm) {
      int m = wm * 32 + sm * 16 + mL;
      af[sm].u[0] = *reinterpret_cast<const uint4*>(&lA[cur][m][khalf]);
      af[sm].u[1] = *reinterpret_cast<const uint4*>(&lA[cur][m][khalf + 8]);
    }
    // B fragments: 32B contiguous per lane straight from global
    union { v16bf v; uint4 u[2]; } bfr[2];
#pragma unroll
    for (int sn = 0; sn < 2; ++sn) {
      const uint4* bg = reinterpret_cast<const uint4*>(wrow[sn] + k0);
      bfr[sn].u[0] = bg[0];
      bfr[sn].u[1] = bg[1];
    }

#pragma unroll
    for (int sm = 0; sm < 2; ++sm)
#pragma unroll
      for (int sn = 0; sn < 2; ++sn)
        acc[sm][sn] = __builtin_amdgcn_wmma_f32_16x16x32_bf16(
            false, af[sm].v, false, bfr[sn].v, (short)0, acc[sm][sn],
            false, false);

    __syncthreads();  // readers done before iteration it+1 overwrites 'cur'
  }

  // C/D layout: VGPR g -> row g (+8 for lanes 16-31), col = lane&15
#pragma unroll
  for (int sm = 0; sm < 2; ++sm)
#pragma unroll
    for (int sn = 0; sn < 2; ++sn) {
      int colBase = blockCol + wn * 32 + sn * 16 + (lane & 15);
      float bv = bias[colBase];
#pragma unroll
      for (int g = 0; g < 8; ++g) {
        long row = blockRow + wm * 32 + sm * 16 + g + 8 * (lane >> 4);
        if (row < M && colBase < Nout)
          C[row * Nout + colBase] = acc[sm][sn][g] + bv;
      }
    }
}

// ------------------------------ edge phase ---------------------------------
// qkvr packed per node: [q | k | v | r], row stride = 4*d

__global__ __launch_bounds__(256) void edge_logits_kernel(
    const float* __restrict__ pack, long stride,
    const int* __restrict__ src, const int* __restrict__ dst,
    float* __restrict__ logits, unsigned* __restrict__ maxkey,
    int E, int d, float scale) {
  int e = blockIdx.x * 8 + (threadIdx.x >> 5);
  if (e >= E) return;
  int lane = threadIdx.x & 31;
  int s = src[e], t = dst[e];
  const float* qp = pack + (long)t * stride;       // q at offset 0
  const float* kp = pack + (long)s * stride + d;   // k at offset d
  float acc = 0.f;
  for (int j = lane; j < d; j += 32) acc += qp[j] * kp[j];
#pragma unroll
  for (int off = 16; off; off >>= 1) acc += __shfl_xor(acc, off, 32);
  if (lane == 0) {
    float lg = acc * scale;
    logits[e] = lg;
    __hip_atomic_fetch_max(&maxkey[t], fkey(lg), __ATOMIC_RELAXED,
                           __HIP_MEMORY_SCOPE_AGENT);
  }
}

__global__ __launch_bounds__(256) void edge_expsum_kernel(
    const float* __restrict__ logits, const unsigned* __restrict__ maxkey,
    const int* __restrict__ dst, float* __restrict__ ex,
    float* __restrict__ denom, int E) {
  int e = blockIdx.x * blockDim.x + threadIdx.x;
  if (e >= E) return;
  int t = dst[e];
  float m = funkey(maxkey[t]);
  float x = __expf(logits[e] - m);
  ex[e] = x;
  atomic_add_f32(&denom[t], x);
}

__global__ __launch_bounds__(256) void edge_scatter_kernel(
    const float* __restrict__ ex, const float* __restrict__ denom,
    const float* __restrict__ pack, long stride,
    const int* __restrict__ src, const int* __restrict__ dst,
    float* __restrict__ attn, int E, int d) {
  int e = blockIdx.x * 8 + (threadIdx.x >> 5);
  if (e >= E) return;
  int lane = threadIdx.x & 31;
  int s = src[e], t = dst[e];
  float alpha = ex[e] / denom[t];
  const float* vp = pack + (long)s * stride + 2 * d;  // v at offset 2d
  float* op = attn + (long)t * d;
  for (int j = lane; j < d; j += 32) atomic_add_f32(&op[j], alpha * vp[j]);
}

// one wave32 per node: beta gate + optional ELU, dual (f32 / bf16) writes
__global__ __launch_bounds__(256) void node_gate_kernel(
    const float* __restrict__ attn,
    const float* __restrict__ pack, long stride,   // r at offset 3d
    const float* __restrict__ Wb,
    float* yf, long fstride, long foff,
    __bf16* ybf, long bstride, long boff,
    int n_nodes, int d, int do_elu) {
  int n = blockIdx.x * 8 + (threadIdx.x >> 5);
  if (n >= n_nodes) return;
  int lane = threadIdx.x & 31;
  const float* op = attn + (long)n * d;
  const float* rp = pack + (long)n * stride + 3 * d;
  float s = 0.f;
  for (int j = lane; j < d; j += 32) {
    float o = op[j], rr = rp[j];
    s += Wb[j] * o + Wb[d + j] * rr + Wb[2 * d + j] * (o - rr);
  }
#pragma unroll
  for (int off = 16; off; off >>= 1) s += __shfl_xor(s, off, 32);
  float beta = 1.f / (1.f + __expf(-s));
  for (int j = lane; j < d; j += 32) {
    float o = op[j], rr = rp[j];
    float y = beta * rr + (1.f - beta) * o;
    if (do_elu) y = (y > 0.f) ? y : expm1f(y);
    if (yf)  yf[(long)n * fstride + foff + j] = y;
    if (ybf) ybf[(long)n * bstride + boff + j] = (__bf16)y;
  }
}

// ------------------------------ host orchestration -------------------------

namespace {

struct Ctx {
  hipStream_t st;
  const int *src, *dst;
  __bf16* wt;        // packed 4x weights bf16 (padded)
  float*  bp;        // packed bias (padded)
  float*  pack;      // qkvr packed [N, 4*dout]
  float*  attn;      // [N, dout]
  float *logits, *ex;
  float* denom;
  unsigned* maxkey;
};

inline size_t alignup(size_t x) { return (x + 255) & ~(size_t)255; }

void run_tconv(const Ctx& c, void* const* d_in, int layer,
               const __bf16* xbf, int ldx, int din, int dout,
               float* yf, long fstride, long foff,
               __bf16* ybf, long bstride, long boff, int do_elu) {
  const float* Wq = (const float*)d_in[3 + layer * 9 + 0];
  const float* bq = (const float*)d_in[3 + layer * 9 + 1];
  const float* Wk = (const float*)d_in[3 + layer * 9 + 2];
  const float* bk = (const float*)d_in[3 + layer * 9 + 3];
  const float* Wv = (const float*)d_in[3 + layer * 9 + 4];
  const float* bv = (const float*)d_in[3 + layer * 9 + 5];
  const float* Ws = (const float*)d_in[3 + layer * 9 + 6];
  const float* bs = (const float*)d_in[3 + layer * 9 + 7];
  const float* Wb = (const float*)d_in[3 + layer * 9 + 8];

  const int Kpad = (din + 31) & ~31;
  const int Jall = 4 * dout;
  const int Jpad = (Jall + 63) & ~63;

  // pack weights + bias
  {
    long total = (long)Jpad * Kpad;
    pack_weights<<<(int)((total + 255) / 256), 256, 0, c.st>>>(
        Wq, Wk, Wv, Ws, c.wt, dout, din, Kpad, total);
    pack_bias<<<(Jpad + 255) / 256, 256, 0, c.st>>>(bq, bk, bv, bs, c.bp,
                                                    dout, Jpad);
  }

  // fused q|k|v|r projection
  dim3 gg(Jpad / 64, (NN + 127) / 128);
  gemm_bf16_wmma<<<gg, 256, 0, c.st>>>(xbf, ldx, c.wt, Kpad, c.bp, c.pack,
                                       NN, Jall, Kpad);

  // attention
  hipMemsetAsync(c.maxkey, 0, (size_t)NN * 4, c.st);
  hipMemsetAsync(c.denom, 0, (size_t)NN * 4, c.st);
  hipMemsetAsync(c.attn, 0, (size_t)NN * dout * 4, c.st);

  float scale = 1.0f / sqrtf((float)dout);
  long stride = Jall;
  edge_logits_kernel<<<(EE + 7) / 8, 256, 0, c.st>>>(
      c.pack, stride, c.src, c.dst, c.logits, c.maxkey, EE, dout, scale);
  edge_expsum_kernel<<<(EE + 255) / 256, 256, 0, c.st>>>(
      c.logits, c.maxkey, c.dst, c.ex, c.denom, EE);
  edge_scatter_kernel<<<(EE + 7) / 8, 256, 0, c.st>>>(
      c.ex, c.denom, c.pack, stride, c.src, c.dst, c.attn, EE, dout);

  node_gate_kernel<<<(NN + 7) / 8, 256, 0, c.st>>>(
      c.attn, c.pack, stride, Wb, yf, fstride, foff, ybf, bstride, boff,
      NN, dout, do_elu);
}

}  // namespace

extern "C" void kernel_launch(void* const* d_in, const int* in_sizes, int n_in,
                              void* d_out, int out_size, void* d_ws,
                              size_t ws_size, hipStream_t stream) {
  (void)in_sizes; (void)n_in; (void)out_size; (void)ws_size;
  const float* features = (const float*)d_in[0];  // [N, 3000]
  const float* img_feat = (const float*)d_in[1];  // [N, 1024]
  const int*   eidx     = (const int*)d_in[2];    // [2, E]
  float* out = (float*)d_out;

  // ---- workspace carve ----
  char* p = (char*)d_ws;
  auto alloc = [&](size_t bytes) {
    char* r = p;
    p += alignup(bytes);
    return r;
  };
  Ctx c;
  c.st  = stream;
  c.src = eidx;
  c.dst = eidx + EE;
  c.pack   = (float*)alloc((size_t)NN * 12000 * 4);   // max 4*dout = 12000
  c.attn   = (float*)alloc((size_t)NN * 3000 * 4);
  c.logits = (float*)alloc((size_t)EE * 4);
  c.ex     = (float*)alloc((size_t)EE * 4);
  c.denom  = (float*)alloc((size_t)NN * 4);
  c.maxkey = (unsigned*)alloc((size_t)NN * 4);
  c.wt     = (__bf16*)alloc((size_t)6160384 * 2 + 65536);  // >= max Jpad*Kpad
  c.bp     = (float*)alloc((size_t)12032 * 4);
  // bf16 activations, padded to MPAD rows (pad rows zeroed below)
  __bf16* feat_bf   = (__bf16*)alloc((size_t)MPAD * 3008 * 2);
  __bf16* img_bf    = (__bf16*)alloc((size_t)MPAD * 1024 * 2);
  __bf16* h1_bf     = (__bf16*)alloc((size_t)MPAD * 512 * 2);
  __bf16* h3_bf     = (__bf16*)alloc((size_t)MPAD * 512 * 2);
  __bf16* i1_bf     = (__bf16*)alloc((size_t)MPAD * 512 * 2);
  __bf16* i3_bf     = (__bf16*)alloc((size_t)MPAD * 512 * 2);
  __bf16* concat_bf = (__bf16*)alloc((size_t)MPAD * 128 * 2);  // [h2 | i2]
  __bf16* comb_bf   = (__bf16*)alloc((size_t)MPAD * 64 * 2);
  __bf16* c2_bf     = (__bf16*)alloc((size_t)MPAD * 64 * 2);
  __bf16* c3_bf     = (__bf16*)alloc((size_t)MPAD * 512 * 2);

  // zero the pad rows of gate-produced activation buffers (read by GEMM A)
  auto zero_padrows = [&](__bf16* buf, int ld) {
    hipMemsetAsync(buf + (size_t)NN * ld, 0,
                   (size_t)(MPAD - NN) * ld * 2, stream);
  };
  zero_padrows(h1_bf, 512);
  zero_padrows(h3_bf, 512);
  zero_padrows(i1_bf, 512);
  zero_padrows(i3_bf, 512);
  zero_padrows(concat_bf, 128);
  zero_padrows(comb_bf, 64);
  zero_padrows(c2_bf, 64);
  zero_padrows(c3_bf, 512);

  // output layout (flat, tuple order: h2, i2, c2, h4, i4, c4)
  const long o_h2 = 0;
  const long o_i2 = (long)NN * 64;
  const long o_c2 = (long)NN * 128;
  const long o_h4 = (long)NN * 192;
  const long o_i4 = o_h4 + (long)NN * 3000;
  const long o_c4 = o_i4 + (long)NN * 1024;

  // cast graph inputs to padded bf16
  {
    long t1 = (long)MPAD * 3008, t2 = (long)MPAD * 1024;
    cast_pad_bf16<<<(int)((t1 + 255) / 256), 256, 0, stream>>>(
        features, feat_bf, NN, 3000, 3008, t1);
    cast_pad_bf16<<<(int)((t2 + 255) / 256), 256, 0, stream>>>(
        img_feat, img_bf, NN, 1024, 1024, t2);
  }

  // ---- feature branch ----
  run_tconv(c, d_in, 0, feat_bf, 3008, 3000, 512,
            nullptr, 0, 0, h1_bf, 512, 0, 1);                 // h1 = elu(conv1)
  run_tconv(c, d_in, 1, h1_bf, 512, 512, 64,
            out + o_h2, 64, 0, concat_bf, 128, 0, 0);         // h2 = conv2
  run_tconv(c, d_in, 2, concat_bf, 128, 64, 512,
            nullptr, 0, 0, h3_bf, 512, 0, 1);                 // h3 = elu(conv3)
  run_tconv(c, d_in, 3, h3_bf, 512, 512, 3000,
            out + o_h4, 3000, 0, nullptr, 0, 0, 0);           // h4 = conv4

  // ---- image branch ----
  run_tconv(c, d_in, 4, img_bf, 1024, 1024, 512,
            nullptr, 0, 0, i1_bf, 512, 0, 1);                 // i1
  run_tconv(c, d_in, 5, i1_bf, 512, 512, 64,
            out + o_i2, 64, 0, concat_bf, 128, 64, 0);        // i2
  run_tconv(c, d_in, 6, concat_bf + 64, 128, 64, 512,
            nullptr, 0, 0, i3_bf, 512, 0, 1);                 // i3
  run_tconv(c, d_in, 7, i3_bf, 512, 512, 1024,
            out + o_i4, 1024, 0, nullptr, 0, 0, 0);           // i4

  // ---- neck ----
  run_tconv(c, d_in, 8, concat_bf, 128, 128, 64,
            nullptr, 0, 0, comb_bf, 64, 0, 1);                // comb = elu(neck)
  run_tconv(c, d_in, 9, comb_bf, 64, 64, 64,
            out + o_c2, 64, 0, c2_bf, 64, 0, 0);              // c2 = neck2
  run_tconv(c, d_in, 10, c2_bf, 64, 64, 512,
            nullptr, 0, 0, c3_bf, 512, 0, 1);                 // c3
  run_tconv(c, d_in, 11, c3_bf, 512, 512, 3000,
            out + o_c4, 3000, 0, nullptr, 0, 0, 0);           // c4
}
